// iNGPFieldMS_17119739642233
// MI455X (gfx1250) — matrix operand order repaired and checked by reading, not compiled, not run on hardware
//
#include <hip/hip_runtime.h>
#include <hip/hip_bf16.h>
#include <math.h>

typedef __attribute__((ext_vector_type(16))) _Float16 v16h;
typedef __attribute__((ext_vector_type(8)))  _Float16 v8h;
typedef __attribute__((ext_vector_type(8)))  float    v8f;

#define N_PTS  524288
#define NF     8
#define HID    128
#define GEO    15
#define APP    32

// workspace int layout (at byte 0):
// ints[0..7]=counts  ints[8..15]=cursor  ints[16..24]=fieldBase  ints[25..33]=tileOff
#define WS_PW_OFF    256                 // packed f16 weights, 256 tiles * 512 halves = 256 KB
#define WS_ORDER_OFF (256 + 256 * 1024)  // order[N_PTS] ints = 2 MB

// ---------------- helpers ----------------

__device__ __forceinline__ int assign_of(const float* __restrict__ pos,
                                         const float* __restrict__ cen, int p) {
  float x = pos[p * 3 + 0], y = pos[p * 3 + 1], z = pos[p * 3 + 2];
  float best = 3.4e38f; int bi = 0;
  #pragma unroll
  for (int i = 0; i < NF; i++) {
    float dx = x - cen[i * 3 + 0];
    float dy = y - cen[i * 3 + 1];
    float dz = z - cen[i * 3 + 2];
    float d = dx * dx + dy * dy + dz * dz;
    if (d < best) { best = d; bi = i; }   // first-min tie-break like jnp.argmin
  }
  return bi;
}

__device__ __forceinline__ v8f splat8(float b) {
  v8f c;
  #pragma unroll
  for (int r = 0; r < 8; r++) c[r] = b;
  return c;
}

// A fragment (16x32 f16, M in lanes) gathered from row-major LDS tile.
// Lane layout: M = lane&15; lanes 0-15 hold K = 0..7,16..23 ; lanes 16-31 hold K = 8..15,24..31.
__device__ __forceinline__ v16h load_a_frag(const _Float16* lds, int strideH, int kbase, int lane) {
  int M = lane & 15;
  int g = lane >> 4;
  const _Float16* p0 = lds + M * strideH + kbase + g * 8;
  v8h lo = *(const v8h*)(p0);
  v8h hi = *(const v8h*)(p0 + 16);
  v16h r;
  #pragma unroll
  for (int i = 0; i < 8; i++) { r[i] = lo[i]; r[8 + i] = hi[i]; }
  return r;
}

// B fragment: pre-packed per-lane contiguous (tile*512 + lane*16 halves)
__device__ __forceinline__ v16h load_b_frag(const _Float16* __restrict__ pw, int tile, int lane) {
  return *(const v16h*)(pw + (size_t)tile * 512 + lane * 16);
}

__device__ __forceinline__ v8f wmma32(v16h a, v16h b, v8f c) {
  return __builtin_amdgcn_wmma_f32_16x16x32_f16(false, a, false, b, (short)0, c, false, false);
}

// ---------------- setup kernels ----------------

__global__ void k_zero(int* mi) {
  int i = threadIdx.x;
  if (i < 16) mi[i] = 0;
}

__global__ void k_count(const float* __restrict__ pos, const float* __restrict__ cen, int* mi) {
  int p = blockIdx.x * blockDim.x + threadIdx.x;
  if (p >= N_PTS) return;
  atomicAdd(&mi[assign_of(pos, cen, p)], 1);
}

__global__ void k_prefix(int* mi) {
  if (blockIdx.x == 0 && threadIdx.x == 0) {
    int fb = 0, to = 0;
    mi[16] = 0; mi[25] = 0;
    for (int i = 0; i < NF; i++) {
      int c = mi[i];
      mi[8 + i] = fb;                 // cursor starts at field base
      fb += c;  mi[16 + i + 1] = fb;  // fieldBase prefix
      to += (c + 15) >> 4;  mi[25 + i + 1] = to;  // tileOff prefix
    }
  }
}

__global__ void k_scatter(const float* __restrict__ pos, const float* __restrict__ cen,
                          int* mi, int* __restrict__ order) {
  int p = blockIdx.x * blockDim.x + threadIdx.x;
  if (p >= N_PTS) return;
  int f = assign_of(pos, cen, p);
  int idx = atomicAdd(&mi[8 + f], 1);
  order[idx] = p;
}

// Pack weights into per-lane B-fragment layout (f32 -> f16).
// B fragment lane layout: N = lane&15; halves h<8 hold K = (lane>>4)*8 + h, h>=8 hold K = 16 + (lane>>4)*8 + (h-8).
// Tiles: [0,64)   d1  (f*8+nb)        K 0..31 (valid K<3),  N = nb*16+n of 128
//        [64,96)  d2  64+f*4+kb       K = kb*32+kk,          N = n (16 outputs)
//        [96,224) c1  96+(f*8+nb)*2+kb K = kb*32+kk (valid<50), N = nb*16+n
//        [224,256)c2  224+f*4+kb      K = kb*32+kk,          N = n (valid<3)
__global__ void k_pack(const float* __restrict__ Wd1, const float* __restrict__ Wd2,
                       const float* __restrict__ Wc1, const float* __restrict__ Wc2,
                       _Float16* __restrict__ pw) {
  int tid = blockIdx.x * blockDim.x + threadIdx.x;   // 0 .. 131071
  if (tid >= 256 * 512) return;
  int tile = tid >> 9;
  int within = tid & 511;
  int lane = within >> 4;
  int h = within & 15;
  int n = lane & 15;
  int kk = (h < 8) ? (((lane >> 4) << 3) + h) : (16 + ((lane >> 4) << 3) + (h - 8));   // 0..31 within the 32-deep K block
  float v = 0.0f;
  if (tile < 64) {
    int f = tile >> 3, nb = tile & 7;
    if (kk < 3) v = Wd1[(f * 3 + kk) * HID + nb * 16 + n];
  } else if (tile < 96) {
    int u = tile - 64; int f = u >> 2, kb = u & 3; int K = kb * 32 + kk;
    v = Wd2[(f * HID + K) * 16 + n];
  } else if (tile < 224) {
    int u = tile - 96; int f = u >> 4; int nb = (u >> 1) & 7; int kb = u & 1;
    int K = kb * 32 + kk;
    if (K < 3 + GEO + APP) v = Wc1[(f * (3 + GEO + APP) + K) * HID + nb * 16 + n];
  } else {
    int u = tile - 224; int f = u >> 2, kb = u & 3; int K = kb * 32 + kk;
    if (n < 3) v = Wc2[(f * HID + K) * 3 + n];
  }
  pw[tid] = (_Float16)v;
}

// ---------------- main WMMA kernel: one wave = one 16-point tile ----------------

__global__ __launch_bounds__(32)
void k_field(const float* __restrict__ positions,
             const float* __restrict__ directions,
             const float* __restrict__ app,
             const float* __restrict__ bd1,
             const float* __restrict__ bd2,
             const float* __restrict__ bc1,
             const float* __restrict__ bc2,
             const int* __restrict__ mi,
             const int* __restrict__ order,
             const _Float16* __restrict__ pw,
             float* __restrict__ out) {
  __shared__ _Float16 sAd1[16 * 32];    // density layer-1 input tile (pos, K padded to 32)
  __shared__ _Float16 sAc1[16 * 64];    // color layer-1 input tile (dir|geo|app, K padded to 64)
  __shared__ _Float16 sH[16 * HID];     // density hidden
  __shared__ _Float16 sHc[16 * HID];    // color hidden
  __shared__ float    sDens[16];
  __shared__ int      sPt[16];

  int w = blockIdx.x;
  if (w >= mi[33]) return;              // tileOff[8] = total tiles
  int f = 0;
  #pragma unroll
  for (int i = 0; i < 7; i++) if (w >= mi[26 + i]) f = i + 1;
  int t    = w - mi[25 + f];
  int cnt  = mi[f];
  int base = mi[16 + f];
  int rem  = cnt - t * 16;              // >= 1 active slots in this tile
  int lane = threadIdx.x;
  int n    = lane & 15;
  int Mb   = (lane >> 4) * 8;           // C/D row base for this lane group

  if (lane < 16) {
    int s = lane;
    int ss = (s < rem) ? s : (rem - 1); // pad with a valid point; padded slots never stored
    sPt[s] = order[base + t * 16 + ss];
  }
  #pragma unroll
  for (int i = 0; i < 8; i++)  ((unsigned int*)sAd1)[lane + i * 32] = 0u;
  #pragma unroll
  for (int i = 0; i < 16; i++) ((unsigned int*)sAc1)[lane + i * 32] = 0u;
  __syncthreads();

  if (lane < 16) {
    int p = sPt[lane];
    #pragma unroll
    for (int c = 0; c < 3; c++) {
      sAd1[lane * 32 + c] = (_Float16)positions[p * 3 + c];
      sAc1[lane * 64 + c] = (_Float16)directions[p * 3 + c];
    }
  }
  // appearance: 32 lanes x 16 points, coalesced 128B rows
  for (int s = 0; s < 16; s++) {
    float v = app[sPt[s] * APP + lane];
    sAc1[s * 64 + 3 + GEO + lane] = (_Float16)v;
  }
  __syncthreads();

  // ---- density layer 1: h = relu(pos @ Wd1 + bd1), 8 WMMAs ----
  v16h a1 = load_a_frag(sAd1, 32, 0, lane);
  #pragma unroll
  for (int nb = 0; nb < 8; nb++) {
    v8f c = splat8(bd1[f * HID + nb * 16 + n]);
    c = wmma32(a1, load_b_frag(pw, f * 8 + nb, lane), c);
    #pragma unroll
    for (int r = 0; r < 8; r++) {
      float v = c[r];
      sH[(Mb + r) * HID + nb * 16 + n] = (_Float16)(v > 0.0f ? v : 0.0f);
    }
  }
  __syncthreads();

  // ---- density layer 2: dout = h @ Wd2 + bd2, 4 chained WMMAs (N=16 = 1 dens + 15 geo) ----
  {
    v8f c = splat8(bd2[f * 16 + n]);
    #pragma unroll
    for (int kb = 0; kb < 4; kb++) {
      v16h a = load_a_frag(sH, HID, kb * 32, lane);
      c = wmma32(a, load_b_frag(pw, 64 + f * 4 + kb, lane), c);
    }
    if (n == 0) {
      #pragma unroll
      for (int r = 0; r < 8; r++) sDens[Mb + r] = expf(c[r]);
    } else {
      #pragma unroll
      for (int r = 0; r < 8; r++) sAc1[(Mb + r) * 64 + 3 + (n - 1)] = (_Float16)c[r];
    }
  }
  __syncthreads();

  // ---- color layer 1: hc = relu([dir|geo|app] @ Wc1 + bc1), 16 WMMAs ----
  v16h ac0 = load_a_frag(sAc1, 64, 0, lane);
  v16h ac1 = load_a_frag(sAc1, 64, 32, lane);
  #pragma unroll
  for (int nb = 0; nb < 8; nb++) {
    v8f c = splat8(bc1[f * HID + nb * 16 + n]);
    c = wmma32(ac0, load_b_frag(pw, 96 + (f * 8 + nb) * 2 + 0, lane), c);
    c = wmma32(ac1, load_b_frag(pw, 96 + (f * 8 + nb) * 2 + 1, lane), c);
    #pragma unroll
    for (int r = 0; r < 8; r++) {
      float v = c[r];
      sHc[(Mb + r) * HID + nb * 16 + n] = (_Float16)(v > 0.0f ? v : 0.0f);
    }
  }
  __syncthreads();

  // ---- color layer 2: rgb = sigmoid(hc @ Wc2 + bc2), 4 chained WMMAs ----
  {
    v8f c = splat8((n < 3) ? bc2[f * 3 + n] : 0.0f);
    #pragma unroll
    for (int kb = 0; kb < 4; kb++) {
      v16h a = load_a_frag(sHc, HID, kb * 32, lane);
      c = wmma32(a, load_b_frag(pw, 224 + f * 4 + kb, lane), c);
    }
    if (n < 3) {
      #pragma unroll
      for (int r = 0; r < 8; r++) {
        int M = Mb + r;
        if (M < rem) out[sPt[M] * 4 + 1 + n] = 1.0f / (1.0f + expf(-c[r]));
      }
    }
    if (lane < 16 && lane < rem) out[sPt[lane] * 4] = sDens[lane];
  }
}

// ---------------- launcher ----------------

extern "C" void kernel_launch(void* const* d_in, const int* in_sizes, int n_in,
                              void* d_out, int out_size, void* d_ws, size_t ws_size,
                              hipStream_t stream) {
  const float* positions  = (const float*)d_in[0];
  const float* directions = (const float*)d_in[1];
  const float* app        = (const float*)d_in[2];
  const float* centroids  = (const float*)d_in[3];
  const float* Wd1        = (const float*)d_in[4];
  const float* bd1        = (const float*)d_in[5];
  const float* Wd2        = (const float*)d_in[6];
  const float* bd2        = (const float*)d_in[7];
  const float* Wc1        = (const float*)d_in[8];
  const float* bc1        = (const float*)d_in[9];
  const float* Wc2        = (const float*)d_in[10];
  const float* bc2        = (const float*)d_in[11];

  int*       mi    = (int*)d_ws;
  _Float16*  pw    = (_Float16*)((char*)d_ws + WS_PW_OFF);
  int*       order = (int*)((char*)d_ws + WS_ORDER_OFF);
  float*     out   = (float*)d_out;

  k_zero   <<<1, 32, 0, stream>>>(mi);
  k_pack   <<<512, 256, 0, stream>>>(Wd1, Wd2, Wc1, Wc2, pw);
  k_count  <<<N_PTS / 256, 256, 0, stream>>>(positions, centroids, mi);
  k_prefix <<<1, 1, 0, stream>>>(mi);
  k_scatter<<<N_PTS / 256, 256, 0, stream>>>(positions, centroids, mi, order);
  k_field  <<<N_PTS / 16 + NF, 32, 0, stream>>>(positions, directions, app,
                                                bd1, bd2, bc1, bc2,
                                                mi, order, pw, out);
  (void)in_sizes; (void)n_in; (void)out_size; (void)ws_size;
}
